// ConditionalRoutingLayer_42288247996510
// MI455X (gfx1250) — matrix operand order repaired and checked by reading, compiled, not run on hardware
//
#include <hip/hip_runtime.h>
#include <hip/hip_bf16.h>
#include <math.h>

// Problem constants (from reference)
#define DI  2048      // model dim D
#define SS  2048      // sequence length S
#define BB  2         // batch
#define TT  (BB*SS)   // 4096 tokens
#define RRK 32        // low-rank R
#define FFD 8192      // FFN dim F
#define HH  16        // heads
#define DHD 128       // head dim
#define TPAD 40       // padded LDS tile row stride (32 data + 8 pad bf16)

typedef __bf16 bf16;
typedef __attribute__((ext_vector_type(16))) __bf16 v16bf;
typedef __attribute__((ext_vector_type(8)))  __bf16 v8bf;
typedef __attribute__((ext_vector_type(8)))  float  v8f;
typedef __attribute__((ext_vector_type(4)))  unsigned int u32x4;
typedef __attribute__((ext_vector_type(8)))  int i32x8;
typedef __attribute__((ext_vector_type(4)))  int i32x4;

#define HAVE_TDM (__has_builtin(__builtin_amdgcn_tensor_load_to_lds))

__device__ __forceinline__ v8f wmma_bf16(v16bf a, v16bf b, v8f c) {
  // D = A(16x32 bf16) x B(32x16 bf16) + C(16x16 f32)
  return __builtin_amdgcn_wmma_f32_16x16x32_bf16(false, a, false, b, (short)0, c,
                                                 false, false);
}

// fp32 -> bf16 round-to-nearest-even via bit manipulation
__device__ __forceinline__ bf16 f2bf(float f) {
  unsigned u = __float_as_uint(f);
  u += 0x7FFFu + ((u >> 16) & 1u);
  unsigned short h = (unsigned short)(u >> 16);
  return __builtin_bit_cast(bf16, h);
}

#if HAVE_TDM
// Issue one TDM 2D tile load: 32 (contig, bf16) x 128 rows, row stride K elems,
// into LDS at ldsOff with 4-DWORD padding every 16 DWORDs (row stride 40 elems).
__device__ __forceinline__ void tdm_load_tile(const bf16* gaddr, unsigned ldsOff,
                                              int K, int rows) {
  unsigned long long ga = (unsigned long long)(size_t)gaddr;
  u32x4 g0;
  g0[0] = 1u;                                            // count=1, user mode
  g0[1] = ldsOff;                                        // lds_addr (bytes)
  g0[2] = (unsigned)(ga & 0xFFFFFFFFu);                  // global_addr[31:0]
  g0[3] = (unsigned)((ga >> 32) & 0x01FFFFFFu)           // global_addr[56:32]
        | 0x80000000u;                                   // type=2 ("image")
  i32x8 g1;
  g1[0] = (int)((1u << 16)      // data_size = 1 -> 2 bytes
              | (1u << 20)      // pad_enable
              | (3u << 22)      // pad_interval = 16 DWORDs
              | (3u << 25));    // pad_amount   = 4 DWORDs
  g1[1] = (int)(((unsigned)K & 0xFFFFu) << 16);          // tensor_dim0 lo16
  g1[2] = (int)(((unsigned)K >> 16)                      // tensor_dim0 hi16
              | (((unsigned)rows & 0xFFFFu) << 16));     // tensor_dim1 lo16
  g1[3] = (int)((((unsigned)rows >> 16) & 0xFFFFu)       // tensor_dim1 hi16
              | (32u << 16));                            // tile_dim0 = 32
  g1[4] = 128;                                           // tile_dim1 = 128
  g1[5] = K;                                             // dim0_stride lo32
  g1[6] = 0;
  g1[7] = 0;
  i32x4 z4 = {0, 0, 0, 0};
#if __clang_major__ >= 23
  i32x8 z8 = {0, 0, 0, 0, 0, 0, 0, 0};
  __builtin_amdgcn_tensor_load_to_lds(g0, g1, z4, z4, z8, 0);
#else
  __builtin_amdgcn_tensor_load_to_lds(g0, g1, z4, z4, 0);
#endif
}
#endif

// ---------------------------------------------------------------------------
// Weight conversion fp32 -> bf16 (bf16 weight set ~100MB -> L2 resident)
// ---------------------------------------------------------------------------
__global__ void cvt_f32_bf16(const float* __restrict__ src, bf16* __restrict__ dst, int n) {
  int i = (blockIdx.x * blockDim.x + threadIdx.x) * 4;
  if (i + 3 < n) {
    float4 f = *(const float4*)(src + i);
    dst[i + 0] = f2bf(f.x); dst[i + 1] = f2bf(f.y);
    dst[i + 2] = f2bf(f.z); dst[i + 3] = f2bf(f.w);
  } else {
    for (; i < n; ++i) dst[i] = f2bf(src[i]);
  }
}

// ---------------------------------------------------------------------------
// Fused: router logits+argmax, path0 (bias), path1 (low-rank), LN1 -> bf16
// ---------------------------------------------------------------------------
__global__ __launch_bounds__(256)
void router_ln1_kernel(const float* __restrict__ x,
                       const float* __restrict__ sbias,
                       const float* __restrict__ w_down,   // [R, D]
                       const float* __restrict__ w_up,     // [D, R]
                       const float* __restrict__ w_router, // [3, D]
                       const float* __restrict__ g1, const float* __restrict__ b1,
                       int* __restrict__ dec, float* __restrict__ out,
                       bf16* __restrict__ xn) {
  const int t = blockIdx.x, tid = threadIdx.x;
  __shared__ float accs[37];              // 0:sum 1:sumsq 2-4:router 5-36:down-proj
  if (tid < 37) accs[tid] = 0.f;
  __syncthreads();
  const float* xr = x + (size_t)t * DI;
  float xl[8];
  float s = 0.f, ss = 0.f, r0 = 0.f, r1 = 0.f, r2 = 0.f;
  float dr[RRK];
#pragma unroll
  for (int r = 0; r < RRK; ++r) dr[r] = 0.f;
#pragma unroll
  for (int i = 0; i < 8; ++i) {
    int d = tid + i * 256;
    float v = xr[d];
    xl[i] = v;
    s += v; ss += v * v;
    r0 += v * w_router[d];
    r1 += v * w_router[DI + d];
    r2 += v * w_router[2 * DI + d];
#pragma unroll
    for (int r = 0; r < RRK; ++r) dr[r] += v * w_down[(size_t)r * DI + d];
  }
  atomicAdd(&accs[0], s);  atomicAdd(&accs[1], ss);
  atomicAdd(&accs[2], r0); atomicAdd(&accs[3], r1); atomicAdd(&accs[4], r2);
#pragma unroll
  for (int r = 0; r < RRK; ++r) atomicAdd(&accs[5 + r], dr[r]);
  __syncthreads();
  const float mu   = accs[0] * (1.f / DI);
  const float var  = accs[1] * (1.f / DI) - mu * mu;
  const float rstd = rsqrtf(var + 1e-5f);
  int d0 = 0; float best = accs[2];
  if (accs[3] > best) { best = accs[3]; d0 = 1; }
  if (accs[4] > best) { best = accs[4]; d0 = 2; }
  if (tid == 0) dec[t] = d0;
#pragma unroll
  for (int i = 0; i < 8; ++i) {
    int d = tid + i * 256;
    float v = xl[i];
    xn[(size_t)t * DI + d] = f2bf((v - mu) * rstd * g1[d] + b1[d]);
    if (d0 == 0) {
      out[(size_t)t * DI + d] = v + sbias[d];
    } else if (d0 == 1) {
      float a = v;
#pragma unroll
      for (int r = 0; r < RRK; ++r) a += accs[5 + r] * w_up[d * RRK + r];
      out[(size_t)t * DI + d] = a;
    }
    // d0==2 rows are written by the masked FF2 epilogue
  }
}

// ---------------------------------------------------------------------------
// LN2: per-token mean/var over x2 (f32) -> bf16 normalized
// ---------------------------------------------------------------------------
__global__ __launch_bounds__(256)
void ln2_kernel(const float* __restrict__ x2, const float* __restrict__ g,
                const float* __restrict__ bta, bf16* __restrict__ xn2) {
  const int t = blockIdx.x, tid = threadIdx.x;
  __shared__ float accs[2];
  if (tid < 2) accs[tid] = 0.f;
  __syncthreads();
  const float* xr = x2 + (size_t)t * DI;
  float xl[8]; float s = 0.f, ss = 0.f;
#pragma unroll
  for (int i = 0; i < 8; ++i) { float v = xr[tid + i * 256]; xl[i] = v; s += v; ss += v * v; }
  atomicAdd(&accs[0], s); atomicAdd(&accs[1], ss);
  __syncthreads();
  const float mu   = accs[0] * (1.f / DI);
  const float rstd = rsqrtf(accs[1] * (1.f / DI) - mu * mu + 1e-5f);
#pragma unroll
  for (int i = 0; i < 8; ++i) {
    int d = tid + i * 256;
    xn2[(size_t)t * DI + d] = f2bf((xl[i] - mu) * rstd * g[d] + bta[d]);
  }
}

// ---------------------------------------------------------------------------
// Tiled WMMA GEMM: C[T x N] = A[T x K] * W[N x K]^T  (bf16 in, f32 accum)
// Block: 256 threads (8 waves), tile 128x128, K-step 32.
// LDS tiles row-major with stride 40 bf16 (bank-conflict-free, matches the
// TDM pad feature: 4 DWORDs pad every 16 DWORDs). Staged via the Tensor Data
// Mover (tensor_load_to_lds + s_wait_tensorcnt) when available, with a manual
// global->LDS fallback otherwise. Double buffered, DMA overlapped with WMMA.
// ---------------------------------------------------------------------------
enum { EPI_QSCALE = 0, EPI_BF16 = 1, EPI_VT = 2, EPI_ORES = 3, EPI_GELU = 4, EPI_FF2 = 5 };

template<int EPI>
__global__ __launch_bounds__(256)
void gemm_bf16_wmma(const bf16* __restrict__ A, const bf16* __restrict__ W,
                    int K, int N,
                    const float* __restrict__ res, const int* __restrict__ dec,
                    float* __restrict__ outF, bf16* __restrict__ outB, float scale) {
  __shared__ bf16 lA[2][128 * TPAD];   // 10KB per buffer
  __shared__ bf16 lB[2][128 * TPAD];
  const int tid  = threadIdx.x;
  const int bm   = blockIdx.y * 128, bn = blockIdx.x * 128;
  const int lane = tid & 31, wv = tid >> 5;
  const int wm   = wv >> 2, wn = wv & 3;       // 2x4 wave grid, each wave 64x32
  const int lh   = lane >> 4, lm = lane & 15;

  const v8f vzero = {0.f, 0.f, 0.f, 0.f, 0.f, 0.f, 0.f, 0.f};
  v8f acc[4][2];
#pragma unroll
  for (int i = 0; i < 4; ++i)
#pragma unroll
    for (int j = 0; j < 2; ++j) acc[i][j] = vzero;

  const bf16* gA = A + (size_t)bm * K;
  const bf16* gB = W + (size_t)bn * K;
  const int nk = K / 32;
  union Uf { v16bf v; v8bf h2[2]; };

#if HAVE_TDM
  if (wv == 0) {
    tdm_load_tile(gA, (unsigned)(size_t)&lA[0][0], K, 128);
    tdm_load_tile(gB, (unsigned)(size_t)&lB[0][0], K, 128);
  }
  for (int kt = 0; kt < nk; ++kt) {
    const int cur = kt & 1;
    if (wv == 0) __builtin_amdgcn_s_wait_tensorcnt(0);
    __syncthreads();                       // buffer `cur` ready for everyone
    if (kt + 1 < nk && wv == 0) {          // DMA next buffer while computing
      tdm_load_tile(gA + (kt + 1) * 32, (unsigned)(size_t)&lA[cur ^ 1][0], K, 128);
      tdm_load_tile(gB + (kt + 1) * 32, (unsigned)(size_t)&lB[cur ^ 1][0], K, 128);
    }
    Uf af[4], bfr[2];
#pragma unroll
    for (int i = 0; i < 4; ++i) {
      const bf16* pa = &lA[cur][((wm * 4 + i) * 16 + lm) * TPAD];
      af[i].h2[0] = *(const v8bf*)(pa + lh * 16);
      af[i].h2[1] = *(const v8bf*)(pa + lh * 16 + 8);
    }
#pragma unroll
    for (int j = 0; j < 2; ++j) {
      const bf16* pb = &lB[cur][((wn * 2 + j) * 16 + lm) * TPAD];
      bfr[j].h2[0] = *(const v8bf*)(pb + lh * 16);
      bfr[j].h2[1] = *(const v8bf*)(pb + lh * 16 + 8);
    }
#pragma unroll
    for (int i = 0; i < 4; ++i)
#pragma unroll
      for (int j = 0; j < 2; ++j)
        acc[i][j] = wmma_bf16(af[i].v, bfr[j].v, acc[i][j]);
  }
#else
  // Manual staging fallback: thread copies one 32B chunk per tile
  const int r = tid >> 1, half = tid & 1;
  const bf16* ga = gA + (size_t)r * K + half * 16;
  const bf16* gb = gB + (size_t)r * K + half * 16;
  auto stage = [&](int buf, int k0) {
    Uf va; va.v = *(const v16bf*)(ga + k0);
    *(v8bf*)&lA[buf][r * TPAD + half * 16]     = va.h2[0];
    *(v8bf*)&lA[buf][r * TPAD + half * 16 + 8] = va.h2[1];
    Uf vb; vb.v = *(const v16bf*)(gb + k0);
    *(v8bf*)&lB[buf][r * TPAD + half * 16]     = vb.h2[0];
    *(v8bf*)&lB[buf][r * TPAD + half * 16 + 8] = vb.h2[1];
  };
  stage(0, 0);
  __syncthreads();
  for (int kt = 0; kt < nk; ++kt) {
    const int cur = kt & 1;
    if (kt + 1 < nk) stage(cur ^ 1, (kt + 1) * 32);
    if (kt + 2 < nk) {
      __builtin_prefetch(ga + (kt + 2) * 32, 0, 0);
      __builtin_prefetch(gb + (kt + 2) * 32, 0, 0);
    }
    Uf af[4], bfr[2];
#pragma unroll
    for (int i = 0; i < 4; ++i) {
      const bf16* pa = &lA[cur][((wm * 4 + i) * 16 + lm) * TPAD];
      af[i].h2[0] = *(const v8bf*)(pa + lh * 16);
      af[i].h2[1] = *(const v8bf*)(pa + lh * 16 + 8);
    }
#pragma unroll
    for (int j = 0; j < 2; ++j) {
      const bf16* pb = &lB[cur][((wn * 2 + j) * 16 + lm) * TPAD];
      bfr[j].h2[0] = *(const v8bf*)(pb + lh * 16);
      bfr[j].h2[1] = *(const v8bf*)(pb + lh * 16 + 8);
    }
#pragma unroll
    for (int i = 0; i < 4; ++i)
#pragma unroll
      for (int j = 0; j < 2; ++j)
        acc[i][j] = wmma_bf16(af[i].v, bfr[j].v, acc[i][j]);
    __syncthreads();
  }
#endif

  // Epilogue: C layout = lane holds col (lane&15), rows rr + 8*(lane>=16)
#pragma unroll
  for (int i = 0; i < 4; ++i) {
#pragma unroll
    for (int j = 0; j < 2; ++j) {
      const int row0 = bm + (wm * 4 + i) * 16 + lh * 8;
      const int col  = bn + (wn * 2 + j) * 16 + lm;
#pragma unroll
      for (int rr = 0; rr < 8; ++rr) {
        const int gm = row0 + rr;
        float v = acc[i][j][rr] * scale;
        if constexpr (EPI == EPI_QSCALE || EPI == EPI_BF16) {
          outB[(size_t)gm * N + col] = f2bf(v);
        } else if constexpr (EPI == EPI_VT) {
          // store V transposed: [b, h, d, s] for contiguous attention B-frags
          const int bb = gm >> 11, si = gm & (SS - 1);
          const int hh = col >> 7, dd = col & (DHD - 1);
          outB[(((size_t)(bb * HH + hh)) * DHD + dd) * SS + si] = f2bf(v);
        } else if constexpr (EPI == EPI_ORES) {
          outF[(size_t)gm * N + col] = res[(size_t)gm * N + col] + v;
        } else if constexpr (EPI == EPI_GELU) {
          float g = 0.5f * v * (1.f + tanhf(0.7978845608028654f *
                                            (v + 0.044715f * v * v * v)));
          outB[(size_t)gm * N + col] = f2bf(g);
        } else { // EPI_FF2: masked routed write of out2
          if (dec[gm] == 2)
            outF[(size_t)gm * N + col] = res[(size_t)gm * N + col] + v;
        }
      }
    }
  }
}

// ---------------------------------------------------------------------------
// Flash attention (causal): one wave per (b, h, 16-query tile).
// Single-wave workgroup => __syncthreads is a legal wave-local LDS fence.
// ---------------------------------------------------------------------------
__global__ __launch_bounds__(32)
void attn_kernel(const bf16* __restrict__ qb,   // [T, D], pre-scaled by 1/sqrt(DH)
                 const bf16* __restrict__ kb,   // [T, D]
                 const bf16* __restrict__ vT,   // [b,h,d,s]
                 bf16* __restrict__ ctx) {      // [T, D]
  __shared__ bf16 ldsP[512];                    // P tile in A-fragment swizzle
  const int lane = threadIdx.x;
  const int lm = lane & 15, lh = lane >> 4;
  const int w  = blockIdx.x;
  const int qt = w & (SS / 16 - 1);
  const int h  = (w >> 7) & (HH - 1);
  const int b  = w >> 11;
  const int tq = b * SS + qt * 16;

  union Uf { v16bf v; v8bf h2[2]; };
  Uf qf[4];
#pragma unroll
  for (int kk = 0; kk < 4; ++kk) {
    const bf16* p = qb + (size_t)(tq + lm) * DI + h * DHD + kk * 32 + lh * 16;
    qf[kk].h2[0] = *(const v8bf*)p;
    qf[kk].h2[1] = *(const v8bf*)(p + 8);
  }

  const v8f vzero = {0.f, 0.f, 0.f, 0.f, 0.f, 0.f, 0.f, 0.f};
  float m_i[8], l_i[8];
  v8f o[8];
#pragma unroll
  for (int rr = 0; rr < 8; ++rr) { m_i[rr] = -3.0e38f; l_i[rr] = 0.f; }
#pragma unroll
  for (int jd = 0; jd < 8; ++jd) o[jd] = vzero;

  const int nch = qt / 2 + 1;                   // 32-key chunks, causal bound
  for (int c = 0; c < nch; ++c) {
    const int kbase = c * 32;
    v8f sc[2];
#pragma unroll
    for (int t2 = 0; t2 < 2; ++t2) {
      v8f s = vzero;
      const int key = kbase + t2 * 16 + lm;
      const bf16* kp = kb + (size_t)(b * SS + key) * DI + h * DHD + lh * 16;
#pragma unroll
      for (int kk = 0; kk < 4; ++kk) {
        v16bf kf = *(const v16bf*)(kp + kk * 32);
        s = wmma_bf16(qf[kk].v, kf, s);
      }
      sc[t2] = s;
    }
    if (c == nch - 1) {                         // diagonal chunk: causal mask
#pragma unroll
      for (int t2 = 0; t2 < 2; ++t2) {
        const int key = kbase + t2 * 16 + lm;
#pragma unroll
        for (int rr = 0; rr < 8; ++rr) {
          const int qrow = qt * 16 + rr + 8 * lh;
          if (key > qrow) sc[t2][rr] = -3.0e38f;
        }
      }
    }
    float alpha[8], p0[8], p1[8];
#pragma unroll
    for (int rr = 0; rr < 8; ++rr) {
      float rmax = fmaxf(sc[0][rr], sc[1][rr]);
#pragma unroll
      for (int off = 1; off < 16; off <<= 1)
        rmax = fmaxf(rmax, __shfl_xor(rmax, off, 32));
      const float mn = fmaxf(m_i[rr], rmax);
      alpha[rr] = __expf(m_i[rr] - mn);
      m_i[rr] = mn;
      p0[rr] = __expf(sc[0][rr] - mn);
      p1[rr] = __expf(sc[1][rr] - mn);
      float rs = p0[rr] + p1[rr];
#pragma unroll
      for (int off = 1; off < 16; off <<= 1)
        rs += __shfl_xor(rs, off, 32);
      l_i[rr] = l_i[rr] * alpha[rr] + rs;
    }
#pragma unroll
    for (int jd = 0; jd < 8; ++jd)
#pragma unroll
      for (int rr = 0; rr < 8; ++rr) o[jd][rr] *= alpha[rr];

    // P (C-layout) -> LDS in A-fragment swizzle
#pragma unroll
    for (int rr = 0; rr < 8; ++rr) {
      const int row   = rr + 8 * lh;
      const int dlane = row;
      const int jj    = lm;
      ldsP[dlane * 16 + jj]        = f2bf(p0[rr]);
      ldsP[(dlane + 16) * 16 + jj] = f2bf(p1[rr]);
    }
    __syncthreads();
    v16bf pf = *(const v16bf*)&ldsP[lane * 16];
    __syncthreads();
#pragma unroll
    for (int jd = 0; jd < 8; ++jd) {
      const bf16* vp = vT + ((size_t)((b * HH + h) * DHD + jd * 16 + lm)) * SS
                          + kbase + lh * 16;
      v16bf vf = *(const v16bf*)vp;
      o[jd] = wmma_bf16(pf, vf, o[jd]);
    }
  }
#pragma unroll
  for (int rr = 0; rr < 8; ++rr) {
    const float inv = 1.f / l_i[rr];
    const int t = tq + rr + 8 * lh;
#pragma unroll
    for (int jd = 0; jd < 8; ++jd)
      ctx[(size_t)t * DI + h * DHD + jd * 16 + lm] = f2bf(o[jd][rr] * inv);
  }
}

// ---------------------------------------------------------------------------
extern "C" void kernel_launch(void* const* d_in, const int* in_sizes, int n_in,
                              void* d_out, int out_size, void* d_ws, size_t ws_size,
                              hipStream_t stream) {
  (void)in_sizes; (void)n_in; (void)out_size; (void)ws_size;
  const float* x        = (const float*)d_in[0];
  const float* sbias    = (const float*)d_in[1];
  const float* w_down   = (const float*)d_in[2];
  const float* w_up     = (const float*)d_in[3];
  const float* w_router = (const float*)d_in[4];
  const float* ln1s     = (const float*)d_in[5];
  const float* ln1b     = (const float*)d_in[6];
  const float* wq       = (const float*)d_in[7];
  const float* wk       = (const float*)d_in[8];
  const float* wv       = (const float*)d_in[9];
  const float* wo       = (const float*)d_in[10];
  const float* ln2s     = (const float*)d_in[11];
  const float* ln2b     = (const float*)d_in[12];
  const float* wff1     = (const float*)d_in[13];
  const float* wff2     = (const float*)d_in[14];
  float* out = (float*)d_out;

  char* ws = (char*)d_ws;
  auto alloc = [&](size_t bytes) {
    char* p = ws; ws += (bytes + 255) & ~(size_t)255; return p;
  };
  int*  dec  = (int*) alloc((size_t)TT * 4);
  bf16* xn   = (bf16*)alloc((size_t)TT * DI * 2);
  bf16* q_b  = (bf16*)alloc((size_t)TT * DI * 2);
  bf16* k_b  = (bf16*)alloc((size_t)TT * DI * 2);
  bf16* vT   = (bf16*)alloc((size_t)TT * DI * 2);
  bf16* ctxb = (bf16*)alloc((size_t)TT * DI * 2);
  float* x2  = (float*)alloc((size_t)TT * DI * 4);
  bf16* xn2  = (bf16*)alloc((size_t)TT * DI * 2);
  bf16* hb   = (bf16*)alloc((size_t)TT * FFD * 2);
  bf16* wqb  = (bf16*)alloc((size_t)DI * DI * 2);
  bf16* wkb  = (bf16*)alloc((size_t)DI * DI * 2);
  bf16* wvb  = (bf16*)alloc((size_t)DI * DI * 2);
  bf16* wob  = (bf16*)alloc((size_t)DI * DI * 2);
  bf16* wf1b = (bf16*)alloc((size_t)DI * FFD * 2);
  bf16* wf2b = (bf16*)alloc((size_t)DI * FFD * 2);

  // 1. Convert weights to bf16 (bf16 weight set ~100MB -> L2 resident)
  const int nDD = DI * DI, nDF = DI * FFD;
  cvt_f32_bf16<<<nDD / 1024, 256, 0, stream>>>(wq,   wqb,  nDD);
  cvt_f32_bf16<<<nDD / 1024, 256, 0, stream>>>(wk,   wkb,  nDD);
  cvt_f32_bf16<<<nDD / 1024, 256, 0, stream>>>(wv,   wvb,  nDD);
  cvt_f32_bf16<<<nDD / 1024, 256, 0, stream>>>(wo,   wob,  nDD);
  cvt_f32_bf16<<<nDF / 1024, 256, 0, stream>>>(wff1, wf1b, nDF);
  cvt_f32_bf16<<<nDF / 1024, 256, 0, stream>>>(wff2, wf2b, nDF);

  // 2. Router + path0/path1 + LN1
  router_ln1_kernel<<<TT, 256, 0, stream>>>(x, sbias, w_down, w_up, w_router,
                                            ln1s, ln1b, dec, out, xn);

  // 3. QKV projections (q pre-scaled by 1/sqrt(DH); v stored transposed)
  const dim3 gQ(DI / 128, TT / 128);
  gemm_bf16_wmma<EPI_QSCALE><<<gQ, 256, 0, stream>>>(xn, wqb, DI, DI,
      nullptr, nullptr, nullptr, q_b, 0.08838834764831845f);
  gemm_bf16_wmma<EPI_BF16><<<gQ, 256, 0, stream>>>(xn, wkb, DI, DI,
      nullptr, nullptr, nullptr, k_b, 1.0f);
  gemm_bf16_wmma<EPI_VT><<<gQ, 256, 0, stream>>>(xn, wvb, DI, DI,
      nullptr, nullptr, nullptr, vT, 1.0f);

  // 4. Causal flash attention
  attn_kernel<<<BB * HH * (SS / 16), 32, 0, stream>>>(q_b, k_b, vT, ctxb);

  // 5. Output projection + residual -> x2 (f32)
  gemm_bf16_wmma<EPI_ORES><<<gQ, 256, 0, stream>>>(ctxb, wob, DI, DI,
      x, nullptr, x2, nullptr, 1.0f);

  // 6. LN2
  ln2_kernel<<<TT, 256, 0, stream>>>(x2, ln2s, ln2b, xn2);

  // 7. FFN up + GELU
  const dim3 gF1(FFD / 128, TT / 128);
  gemm_bf16_wmma<EPI_GELU><<<gF1, 256, 0, stream>>>(xn2, wf1b, DI, FFD,
      nullptr, nullptr, nullptr, hb, 1.0f);

  // 8. FFN down + residual, masked write for route-2 tokens
  gemm_bf16_wmma<EPI_FF2><<<gQ, 256, 0, stream>>>(hb, wf2b, FFD, DI,
      x2, dec, out, nullptr, 1.0f);
}
